// CausalSelfAttention_21835613733582
// MI455X (gfx1250) — matrix operand order repaired and checked by reading, not compiled, not run on hardware
//
#include <hip/hip_runtime.h>
#include <hip/hip_bf16.h>
#include <stdint.h>

#define N_EMBED  1024
#define N_HEADS  16
#define HEAD_DIM 64
#define T_SEQ    2048
#define BATCH    4
#define M_ROWS   (BATCH * T_SEQ)   // 8192
#define NQKV     (3 * N_EMBED)     // 3072

typedef __attribute__((ext_vector_type(16))) __bf16 v16bf;
typedef __attribute__((ext_vector_type(2)))  __bf16 v2bf;
typedef __attribute__((ext_vector_type(8)))  float  v8f;

__device__ inline v8f v8f_zero() {
    v8f z;
#pragma unroll
    for (int i = 0; i < 8; ++i) z[i] = 0.0f;
    return z;
}

// Pack two f32 -> packed bf16 pair (lo in [15:0], hi in [31:16]); lowers to
// v_cvt_pk_bf16_f32 on gfx1250.
__device__ inline uint32_t pack2bf(float lo, float hi) {
    union { v2bf v; uint32_t u; } p;
    p.v[0] = (__bf16)lo;
    p.v[1] = (__bf16)hi;
    return p.u;
}

__device__ inline v8f wmma_bf16(v16bf a, v16bf b, v8f c) {
    // D = A(16x32 bf16) * B(32x16 bf16) + C(16x16 f32)
    return __builtin_amdgcn_wmma_f32_16x16x32_bf16(
        false, a, false, b, (short)0, c, false, false);
}

// A fragment (16x32 bf16). src: LDS row-major [rows][ld], row = lane&15.
// VGPR i holds pair (k0,k0+1), k0 = (i<4?0:16) + (i&3)*2 + (lane>>4)*8.
__device__ inline v16bf load_frag_a(const __bf16* src, int ld) {
    const int lane = threadIdx.x & 31;
    const int m = lane & 15;
    const int h = lane >> 4;
    union { v16bf v; uint32_t u[8]; } f;
    const uint32_t* row = (const uint32_t*)(src + m * ld);
#pragma unroll
    for (int i = 0; i < 8; ++i) {
        const int k0 = ((i < 4) ? 0 : 16) + (i & 3) * 2 + h * 8;
        f.u[i] = row[k0 >> 1];
    }
    return f.v;
}

// B fragment (32x16 bf16). srcN: LDS N-major [cols][ld] (k contiguous), n = lane&15.
// VGPR i holds pair (k0,k0+1), k0 = (i<4?0:16) + (i&3)*2 + (lane>>4)*8.
__device__ inline v16bf load_frag_b(const __bf16* srcN, int ld) {
    const int lane = threadIdx.x & 31;
    const int n = lane & 15;
    const int h = lane >> 4;
    union { v16bf v; uint32_t u[8]; } f;
    const uint32_t* col = (const uint32_t*)(srcN + n * ld);
#pragma unroll
    for (int i = 0; i < 8; ++i) {
        const int k0 = ((i < 4) ? 0 : 16) + (i & 3) * 2 + h * 8;
        f.u[i] = col[k0 >> 1];
    }
    return f.v;
}

// ---------------------------------------------------------------------------
// Kernel 1: qkv = bf16( x[f32, MxK] @ W_attn[f32, KxN] + b_attn )   N = 3072
// Block tile 128x128, BK=32, 256 thr = 8 waves, wave tile 32x64 (2x4 WMMA).
// ---------------------------------------------------------------------------
__global__ __launch_bounds__(256) void qkv_gemm_kernel(
    const float* __restrict__ X, const float* __restrict__ W,
    const float* __restrict__ bias, __bf16* __restrict__ out)
{
    __shared__ __bf16 As[128 * 32];   // [m][k]
    __shared__ __bf16 Bs[128 * 32];   // [n][k]  (transposed)

    const int tid  = threadIdx.x;
    const int lane = tid & 31;
    const int wave = tid >> 5;
    const int wm = wave & 3;          // 0..3 : 32-row slice
    const int wn = wave >> 2;         // 0..1 : 64-col slice
    const int nBase = blockIdx.x * 128;
    const int mBase = blockIdx.y * 128;

    v8f acc[2][4];
#pragma unroll
    for (int i = 0; i < 2; ++i)
#pragma unroll
        for (int j = 0; j < 4; ++j) acc[i][j] = v8f_zero();

    const int arow = tid >> 1;            // 0..127
    const int acol = (tid & 1) * 16;      // 0 / 16
    const int bkp  = tid >> 4;            // 0..15 : k-pair
    const int bn0  = (tid & 15) * 8;      // 0..120

    const float* aG = X + (size_t)(mBase + arow) * N_EMBED + acol;
    const float* bG0 = W + (size_t)(2 * bkp)     * NQKV + nBase + bn0;
    const float* bG1 = W + (size_t)(2 * bkp + 1) * NQKV + nBase + bn0;

    for (int kk = 0; kk < N_EMBED; kk += 32) {
        if (kk + 32 < N_EMBED) {   // pull next tiles toward L2 while we compute
            __builtin_prefetch(aG + kk + 32, 0, 1);
            __builtin_prefetch(bG0 + (size_t)32 * NQKV + kk * 0, 0, 1);
        }
        // A tile: 16 contiguous f32 -> 8 packed bf16 pairs -> 2 x b128 stores
        {
            const float4* ag = (const float4*)(aG + kk);
            float4 a0 = ag[0], a1 = ag[1], a2 = ag[2], a3 = ag[3];
            uint32_t pk[8] = {
                pack2bf(a0.x, a0.y), pack2bf(a0.z, a0.w),
                pack2bf(a1.x, a1.y), pack2bf(a1.z, a1.w),
                pack2bf(a2.x, a2.y), pack2bf(a2.z, a2.w),
                pack2bf(a3.x, a3.y), pack2bf(a3.z, a3.w)};
            uint32_t* ad = (uint32_t*)(As + arow * 32 + acol);
#pragma unroll
            for (int j = 0; j < 8; ++j) ad[j] = pk[j];
        }
        // B tile: two k-rows x 8 n -> packed k-pairs -> 8 x b32 stores [n][k]
        {
            const float4* b0p = (const float4*)(bG0 + (size_t)kk * NQKV);
            const float4* b1p = (const float4*)(bG1 + (size_t)kk * NQKV);
            float4 r0a = b0p[0], r0b = b0p[1];
            float4 r1a = b1p[0], r1b = b1p[1];
            float lo[8] = {r0a.x,r0a.y,r0a.z,r0a.w, r0b.x,r0b.y,r0b.z,r0b.w};
            float hi[8] = {r1a.x,r1a.y,r1a.z,r1a.w, r1b.x,r1b.y,r1b.z,r1b.w};
#pragma unroll
            for (int j = 0; j < 8; ++j)
                *(uint32_t*)(Bs + (bn0 + j) * 32 + 2 * bkp) = pack2bf(lo[j], hi[j]);
        }
        __syncthreads();

        v16bf a0 = load_frag_a(As + (wm * 32)      * 32, 32);
        v16bf a1 = load_frag_a(As + (wm * 32 + 16) * 32, 32);
#pragma unroll
        for (int j = 0; j < 4; ++j) {
            v16bf b = load_frag_b(Bs + (wn * 64 + j * 16) * 32, 32);
            acc[0][j] = wmma_bf16(a0, b, acc[0][j]);
            acc[1][j] = wmma_bf16(a1, b, acc[1][j]);
        }
        __syncthreads();
    }

    const int n0 = lane & 15;
    const int h  = lane >> 4;
#pragma unroll
    for (int si = 0; si < 2; ++si)
#pragma unroll
        for (int sj = 0; sj < 4; ++sj) {
            const int col = nBase + wn * 64 + sj * 16 + n0;
            const float bv = bias[col];
#pragma unroll
            for (int r = 0; r < 8; ++r) {
                const int row = mBase + wm * 32 + si * 16 + r + 8 * h;
                out[(size_t)row * NQKV + col] = (__bf16)(acc[si][sj][r] + bv);
            }
        }
}

// ---------------------------------------------------------------------------
// Kernel 2: flash attention. One block = (b, h, 128-row q tile), 256 thr = 8
// waves, each wave owns 16 q rows. k tiles of 64 keys, kt = 0..2*qt+1 (causal).
// ---------------------------------------------------------------------------
__global__ __launch_bounds__(256) void attn_kernel(
    const __bf16* __restrict__ qkv, __bf16* __restrict__ y)
{
    __shared__ __bf16 Qs[128 * 64];       // [qrow][d]
    __shared__ __bf16 Ks[64 * 64];        // [key][d]
    __shared__ __bf16 Vs[64 * 64];        // [d][key]  (transposed)
    __shared__ __bf16 Ps[8 * 16 * 64];    // per-wave P tile [16][64]

    const int tid  = threadIdx.x;
    const int lane = tid & 31;
    const int wave = tid >> 5;
    const int qt = blockIdx.x & 15;       // q tile (T/128 = 16)
    const int bh = blockIdx.x >> 4;
    const int b  = bh >> 4;
    const int h  = bh & 15;

    const __bf16* base = qkv + ((size_t)b * T_SEQ) * NQKV + h * HEAD_DIM;

    // Q tile -> LDS (32 bf16 per thread)
    {
        const int r = tid >> 1, half = (tid & 1) * 32;
        const uint4* src = (const uint4*)(base + (size_t)(qt * 128 + r) * NQKV + half);
        uint4* dst = (uint4*)(Qs + r * 64 + half);
        dst[0] = src[0]; dst[1] = src[1]; dst[2] = src[2]; dst[3] = src[3];
    }

    const int n0 = lane & 15;
    const int hh = lane >> 4;

    v8f o[4];
    float m[8], l[8];
#pragma unroll
    for (int j = 0; j < 4; ++j) o[j] = v8f_zero();
#pragma unroll
    for (int r = 0; r < 8; ++r) { m[r] = -3.0e38f; l[r] = 0.0f; }

    const int ktEnd = 2 * qt + 1;
    for (int kt = 0; kt <= ktEnd; ++kt) {
        __syncthreads();   // previous iteration's reads of Ks/Vs are done
        // K tile [key][d] (matches B-fragment for S = Q @ K^T)
        {
            const int r = tid >> 2, q4 = (tid & 3) * 16;
            const uint4* src = (const uint4*)(base + N_EMBED +
                                (size_t)(kt * 64 + r) * NQKV + q4);
            uint4* dst = (uint4*)(Ks + r * 64 + q4);
            dst[0] = src[0]; dst[1] = src[1];
        }
        // V tile transposed [d][key]: read key pairs, store packed b32
        {
            const int kp = tid >> 3;            // 0..31 key pair
            const int d0 = (tid & 7) * 8;       // 0..56
            const uint4* s0 = (const uint4*)(base + 2 * N_EMBED +
                                (size_t)(kt * 64 + 2 * kp)     * NQKV + d0);
            const uint4* s1 = (const uint4*)(base + 2 * N_EMBED +
                                (size_t)(kt * 64 + 2 * kp + 1) * NQKV + d0);
            uint4 t0 = s0[0], t1 = s1[0];
            const __bf16* v0 = (const __bf16*)&t0;
            const __bf16* v1 = (const __bf16*)&t1;
#pragma unroll
            for (int d = 0; d < 8; ++d) {
                union { v2bf v; uint32_t u; } p;
                p.v[0] = v0[d]; p.v[1] = v1[d];
                *(uint32_t*)(Vs + (d0 + d) * 64 + 2 * kp) = p.u;
            }
        }
        __syncthreads();

        // S = (Q @ K^T) * 1/sqrt(D) : 16x64 per wave, 8 WMMAs
        v16bf aq0 = load_frag_a(Qs + (wave * 16) * 64,      64);  // d 0..31
        v16bf aq1 = load_frag_a(Qs + (wave * 16) * 64 + 32, 64);  // d 32..63
        v8f s[4];
#pragma unroll
        for (int j = 0; j < 4; ++j) {
            v16bf bk0 = load_frag_b(Ks + (j * 16) * 64,      64);
            v16bf bk1 = load_frag_b(Ks + (j * 16) * 64 + 32, 64);
            s[j] = wmma_bf16(aq0, bk0, v8f_zero());
            s[j] = wmma_bf16(aq1, bk1, s[j]);
#pragma unroll
            for (int r = 0; r < 8; ++r) s[j][r] *= 0.125f;  // 1/sqrt(64)
        }

        // causal mask: only the last two k tiles can cross the diagonal
        if (kt >= 2 * qt) {
#pragma unroll
            for (int j = 0; j < 4; ++j) {
                const int kj = kt * 64 + j * 16 + n0;
#pragma unroll
                for (int r = 0; r < 8; ++r) {
                    const int qi = qt * 128 + wave * 16 + r + 8 * hh;
                    if (kj > qi) s[j][r] = -3.0e38f;
                }
            }
        }

        // online softmax (row r+8*hh lives across the 16 lanes of a half-wave)
#pragma unroll
        for (int r = 0; r < 8; ++r) {
            float smax = fmaxf(fmaxf(s[0][r], s[1][r]), fmaxf(s[2][r], s[3][r]));
#pragma unroll
            for (int off = 1; off < 16; off <<= 1)
                smax = fmaxf(smax, __shfl_xor(smax, off, 32));
            const float mnew = fmaxf(m[r], smax);
            const float corr = __expf(m[r] - mnew);
            float lsum = 0.0f;
#pragma unroll
            for (int j = 0; j < 4; ++j) {
                const float p = __expf(s[j][r] - mnew);
                s[j][r] = p;
                lsum += p;
            }
#pragma unroll
            for (int off = 1; off < 16; off <<= 1)
                lsum += __shfl_xor(lsum, off, 32);
            l[r] = l[r] * corr + lsum;
            m[r] = mnew;
#pragma unroll
            for (int j = 0; j < 4; ++j) o[j][r] *= corr;
        }

        // P (C-layout f32) -> LDS bf16 -> A-layout fragments
        __bf16* pw = Ps + wave * 16 * 64;
#pragma unroll
        for (int j = 0; j < 4; ++j)
#pragma unroll
            for (int r = 0; r < 8; ++r)
                pw[(r + 8 * hh) * 64 + j * 16 + n0] = (__bf16)s[j][r];
        __syncthreads();

        v16bf ap0 = load_frag_a(pw,      64);  // keys 0..31
        v16bf ap1 = load_frag_a(pw + 32, 64);  // keys 32..63
#pragma unroll
        for (int j = 0; j < 4; ++j) {
            v16bf bv0 = load_frag_b(Vs + (j * 16) * 64,      64);
            v16bf bv1 = load_frag_b(Vs + (j * 16) * 64 + 32, 64);
            o[j] = wmma_bf16(ap0, bv0, o[j]);
            o[j] = wmma_bf16(ap1, bv1, o[j]);
        }
    }

    // y[b, t, h*64 + d] = O / l   (already head-merged layout)
#pragma unroll
    for (int r = 0; r < 8; ++r) {
        const int row = b * T_SEQ + qt * 128 + wave * 16 + r + 8 * hh;
        const float inv = 1.0f / l[r];
#pragma unroll
        for (int j = 0; j < 4; ++j) {
            const int col = h * HEAD_DIM + j * 16 + n0;
            y[(size_t)row * N_EMBED + col] = (__bf16)(o[j][r] * inv);
        }
    }
}

// ---------------------------------------------------------------------------
// Kernel 3: out = f32( y[bf16, MxK] @ W_proj[f32, KxN] + b_proj )   N = 1024
// ---------------------------------------------------------------------------
__global__ __launch_bounds__(256) void proj_gemm_kernel(
    const __bf16* __restrict__ Y, const float* __restrict__ W,
    const float* __restrict__ bias, float* __restrict__ out)
{
    __shared__ __bf16 As[128 * 32];
    __shared__ __bf16 Bs[128 * 32];

    const int tid  = threadIdx.x;
    const int lane = tid & 31;
    const int wave = tid >> 5;
    const int wm = wave & 3;
    const int wn = wave >> 2;
    const int nBase = blockIdx.x * 128;
    const int mBase = blockIdx.y * 128;

    v8f acc[2][4];
#pragma unroll
    for (int i = 0; i < 2; ++i)
#pragma unroll
        for (int j = 0; j < 4; ++j) acc[i][j] = v8f_zero();

    const int arow = tid >> 1;
    const int acol = (tid & 1) * 16;
    const int bkp  = tid >> 4;
    const int bn0  = (tid & 15) * 8;

    const __bf16* aG = Y + (size_t)(mBase + arow) * N_EMBED + acol;
    const float* bG0 = W + (size_t)(2 * bkp)     * N_EMBED + nBase + bn0;
    const float* bG1 = W + (size_t)(2 * bkp + 1) * N_EMBED + nBase + bn0;

    for (int kk = 0; kk < N_EMBED; kk += 32) {
        if (kk + 32 < N_EMBED) __builtin_prefetch(aG + kk + 32, 0, 1);
        // A tile already bf16: 2 x b128 copy
        {
            const uint4* ag = (const uint4*)(aG + kk);
            uint4* ad = (uint4*)(As + arow * 32 + acol);
            ad[0] = ag[0]; ad[1] = ag[1];
        }
        // B tile f32 -> packed bf16 k-pairs -> [n][k]
        {
            const float4* b0p = (const float4*)(bG0 + (size_t)kk * N_EMBED);
            const float4* b1p = (const float4*)(bG1 + (size_t)kk * N_EMBED);
            float4 r0a = b0p[0], r0b = b0p[1];
            float4 r1a = b1p[0], r1b = b1p[1];
            float lo[8] = {r0a.x,r0a.y,r0a.z,r0a.w, r0b.x,r0b.y,r0b.z,r0b.w};
            float hi[8] = {r1a.x,r1a.y,r1a.z,r1a.w, r1b.x,r1b.y,r1b.z,r1b.w};
#pragma unroll
            for (int j = 0; j < 8; ++j)
                *(uint32_t*)(Bs + (bn0 + j) * 32 + 2 * bkp) = pack2bf(lo[j], hi[j]);
        }
        __syncthreads();

        v16bf a0 = load_frag_a(As + (wm * 32)      * 32, 32);
        v16bf a1 = load_frag_a(As + (wm * 32 + 16) * 32, 32);
#pragma unroll
        for (int j = 0; j < 4; ++j) {
            v16bf bfr = load_frag_b(Bs + (wn * 64 + j * 16) * 32, 32);
            acc[0][j] = wmma_bf16(a0, bfr, acc[0][j]);
            acc[1][j] = wmma_bf16(a1, bfr, acc[1][j]);
        }
        __syncthreads();
    }

    const int n0 = lane & 15;
    const int h  = lane >> 4;
#pragma unroll
    for (int si = 0; si < 2; ++si)
#pragma unroll
        for (int sj = 0; sj < 4; ++sj) {
            const int col = nBase + wn * 64 + sj * 16 + n0;
            const float bv = bias[col];
#pragma unroll
            for (int r = 0; r < 8; ++r) {
                const int row = mBase + wm * 32 + si * 16 + r + 8 * h;
                out[(size_t)row * N_EMBED + col] = acc[si][sj][r] + bv;
            }
        }
}

// ---------------------------------------------------------------------------
extern "C" void kernel_launch(void* const* d_in, const int* in_sizes, int n_in,
                              void* d_out, int out_size, void* d_ws, size_t ws_size,
                              hipStream_t stream) {
    const float* x      = (const float*)d_in[0];
    const float* W_attn = (const float*)d_in[1];
    const float* b_attn = (const float*)d_in[2];
    const float* W_proj = (const float*)d_in[3];
    const float* b_proj = (const float*)d_in[4];
    float* out = (float*)d_out;

    // workspace: qkv bf16 (48 MB) | y bf16 (16 MB)
    __bf16* qkv = (__bf16*)d_ws;
    __bf16* y   = (__bf16*)((char*)d_ws + (size_t)M_ROWS * NQKV * sizeof(__bf16));

    qkv_gemm_kernel<<<dim3(NQKV / 128, M_ROWS / 128), 256, 0, stream>>>(x, W_attn, b_attn, qkv);
    attn_kernel<<<dim3(BATCH * N_HEADS * (T_SEQ / 128)), 256, 0, stream>>>(qkv, y);
    proj_gemm_kernel<<<dim3(N_EMBED / 128, M_ROWS / 128), 256, 0, stream>>>(y, W_proj, b_proj, out);
}
